// ThreeLayerAttention_24644522344714
// MI455X (gfx1250) — compile-verified, matrix operand order checked by reading, not run on hardware
//
#include <hip/hip_runtime.h>
#include <hip/hip_bf16.h>

// ---------------------------------------------------------------------------
// ThreeLayerAttention on MI455X (gfx1250), bf16 WMMA path, fully padded
// layouts (DPAD=1088) so all GEMM/attention inner loops are guard-free.
// GEMM: 64x64 block tile, double-buffered LDS, 2 WMMA per wave per K-step,
// optional transposed epilogue (used to produce V^T for the P.V phase).
//   B=32, S=1024, D=1025, QD=KD=64.
// ---------------------------------------------------------------------------

typedef __attribute__((ext_vector_type(16))) __bf16 v16bf;
typedef __attribute__((ext_vector_type(8)))  float  v8f;

#define BATCH 32
#define SEQ   1024
#define DIM   1025
#define DPAD  1088            // 17*64: rows 16B-aligned, N%64==0, K%32==0
#define QKD   64
#define NB    (BATCH * SEQ)   // 32768 rows with batch folded into M

// Per-lane K offset for the 16-bit A/B WMMA fragment layout (ISA 7.12.2):
// lane = half*16 + idx; VGPR j holds the bf16 pair at K = base, base+1.
__device__ __forceinline__ int wmma_kb(int j, int hf) {
    return ((j & 4) ? 16 : 0) + ((j & 3) << 1) + (hf << 3);
}

union Frag16 { unsigned int u[8]; v16bf v; };
union Pack4  { unsigned long long q; __bf16 h[4]; };
union Pack8  { uint4 q; __bf16 h[8]; };

// ---------------------------------------------------------------------------
// f32 [srows x scols] (contiguous) -> bf16 [drows x dstride], zero padding
// ---------------------------------------------------------------------------
__global__ void k_cvt_pad(const float* __restrict__ src,
                          __bf16* __restrict__ dst,
                          int srows, int scols, int drows, int dstride) {
    const long long i = (long long)blockIdx.x * blockDim.x + threadIdx.x;
    const long long total = (long long)drows * dstride;
    if (i >= total) return;
    const int r = (int)(i / dstride);
    const int c = (int)(i % dstride);
    dst[i] = (r < srows && c < scols) ? (__bf16)src[(long long)r * scols + c]
                                      : (__bf16)0.0f;
}

// y[:, 1024] = x[:, 1024]; zero y[:, 1025..DPAD)
__global__ void k_fix_lastcols(const __bf16* __restrict__ xb,
                               __bf16* __restrict__ y) {
    const long long i = (long long)blockIdx.x * blockDim.x + threadIdx.x;
    const int r = (int)(i >> 6);                 // DPAD-DIM+1 == 64
    const int c = (int)(i & 63);                 // cols 1024..1087
    if (r >= NB) return;
    const size_t p = (size_t)r * DPAD + (DIM - 1) + c;
    y[p] = (c == 0) ? xb[p] : (__bf16)0.0f;
}

// ---------------------------------------------------------------------------
// Guard-free bf16 GEMM:  C = A[M,K] * B[K,N]   (f32 accum, bf16 out)
// Requires M%64==0, N%64==0, K%32==0, lda%8==0, ldb%8==0.
// Block tile 64x64, 8 waves: wave = (m16, n32); each wave computes two 16x16
// accumulators sharing one A fragment. Double-buffered LDS, 1 barrier/K-step.
// transC==0: C[M,N] row-major (ldc = row stride).
// transC==1: C[N,M] (ldc = row stride = M); lane's 8 accum values are M-
//            consecutive -> one aligned 16B store per accumulator.
// ---------------------------------------------------------------------------
__global__ __launch_bounds__(256) void k_gemm_bf16(
        const __bf16* __restrict__ A, int lda,
        const __bf16* __restrict__ Bm, int ldb,
        __bf16* __restrict__ C, int ldc,
        int K, int transC) {
    __shared__ __bf16 As[2][64][32];     // [buf][m][k]
    __shared__ __bf16 Bts[2][64][32];    // [buf][n][k] (K pairs contiguous)

    const int t    = threadIdx.x;
    const int wave = t >> 5;
    const int lane = t & 31;
    const int hf   = lane >> 4;
    const int idx  = lane & 15;
    const int m16  = wave & 3;           // 0..3  (16-row slab)
    const int n32  = wave >> 2;          // 0..1  (32-col slab)
    const int m0   = blockIdx.x * 64;
    const int n0   = blockIdx.y * 64;

    const int ar  = t >> 2;              // 0..63  A tile row
    const int akc = (t & 3) << 3;        // 0,8,16,24
    const int bk  = t & 31;              // 0..31  B tile k
    const int bn8 = (t >> 5) << 3;       // 0,8,...,56  B tile n base

    const __bf16* aPtr = A + (size_t)(m0 + ar) * lda + akc;      // 16B aligned
    const __bf16* bPtr = Bm + (size_t)bk * ldb + n0 + bn8;       // 8B aligned
    const size_t  bStep = (size_t)32 * ldb;

    const int steps = K >> 5;

    // ---- prologue: stage tile 0 into buffer 0 ----
    uint4 aReg = *reinterpret_cast<const uint4*>(aPtr);
    Pack4 b0, b1;
    b0.q = *reinterpret_cast<const unsigned long long*>(bPtr);
    b1.q = *reinterpret_cast<const unsigned long long*>(bPtr + 4);
    *reinterpret_cast<uint4*>(&As[0][ar][akc]) = aReg;
#pragma unroll
    for (int i = 0; i < 4; ++i) {
        Bts[0][bn8 + i][bk]     = b0.h[i];
        Bts[0][bn8 + 4 + i][bk] = b1.h[i];
    }
    __syncthreads();

    v8f acc0 = {}, acc1 = {};

    for (int i = 0; i < steps; ++i) {
        const int  cur  = i & 1;
        const int  nxt  = cur ^ 1;
        const bool more = (i + 1) < steps;

        // issue next tile's global loads before consuming the current tile
        if (more) {
            const __bf16* ap = aPtr + (size_t)(i + 1) * 32;
            const __bf16* bp = bPtr + (size_t)(i + 1) * bStep;
            __builtin_prefetch(ap + 32, 0, 1);
            aReg = *reinterpret_cast<const uint4*>(ap);
            b0.q = *reinterpret_cast<const unsigned long long*>(bp);
            b1.q = *reinterpret_cast<const unsigned long long*>(bp + 4);
        }

        Frag16 fa, fb0, fb1;
#pragma unroll
        for (int j = 0; j < 8; ++j) {
            const int kb = wmma_kb(j, hf);
            fa.u[j]  = *reinterpret_cast<const unsigned int*>(&As[cur][m16 * 16 + idx][kb]);
            fb0.u[j] = *reinterpret_cast<const unsigned int*>(&Bts[cur][n32 * 32 + idx][kb]);
            fb1.u[j] = *reinterpret_cast<const unsigned int*>(&Bts[cur][n32 * 32 + 16 + idx][kb]);
        }
        acc0 = __builtin_amdgcn_wmma_f32_16x16x32_bf16(
                   false, fa.v, false, fb0.v, (short)0, acc0, false, false);
        acc1 = __builtin_amdgcn_wmma_f32_16x16x32_bf16(
                   false, fa.v, false, fb1.v, (short)0, acc1, false, false);

        if (more) {
            *reinterpret_cast<uint4*>(&As[nxt][ar][akc]) = aReg;
#pragma unroll
            for (int ii = 0; ii < 4; ++ii) {
                Bts[nxt][bn8 + ii][bk]     = b0.h[ii];
                Bts[nxt][bn8 + 4 + ii][bk] = b1.h[ii];
            }
        }
        __syncthreads();   // writes go to nxt, reads were from cur: one barrier
    }

    const int colg0 = n0 + n32 * 32 + idx;
    if (transC) {
        // C[N,M]: lane's 8 values are consecutive in M -> 16B vector stores
        const size_t mbase = (size_t)(m0 + m16 * 16 + hf * 8);
        Pack8 p0, p1;
#pragma unroll
        for (int r = 0; r < 8; ++r) {
            p0.h[r] = (__bf16)acc0[r];
            p1.h[r] = (__bf16)acc1[r];
        }
        *reinterpret_cast<uint4*>(C + (size_t)colg0 * ldc + mbase)        = p0.q;
        *reinterpret_cast<uint4*>(C + (size_t)(colg0 + 16) * ldc + mbase) = p1.q;
    } else {
#pragma unroll
        for (int r = 0; r < 8; ++r) {
            const size_t row = (size_t)(m0 + m16 * 16 + r + hf * 8) * ldc;
            C[row + colg0]      = (__bf16)acc0[r];
            C[row + colg0 + 16] = (__bf16)acc1[r];
        }
    }
}

// ---------------------------------------------------------------------------
// Fused attention, per (batch, 16-query tile):
//   scores = Q(16x64).K^T -> bf16 LDS; row softmax (1/sum folded into store);
//   out(16x1088) = P.V with V supplied transposed: vt[DPAD][NB].
// ---------------------------------------------------------------------------
__global__ __launch_bounds__(256) void k_attention(
        const __bf16* __restrict__ q,    // [NB][64]
        const __bf16* __restrict__ km,   // [NB][64]
        const __bf16* __restrict__ vt,   // [DPAD][NB]  (V transposed)
        __bf16* __restrict__ out) {      // [NB][DPAD]
    __shared__ __bf16 p_lds[16][SEQ];    // 32 KB: raw scores, then exp(P)
    __shared__ __bf16 q_lds[16][QKD];    // 2 KB
    __shared__ float  red[16][16];
    __shared__ float  rowmax[16];
    __shared__ float  invsum[16];

    const int t    = threadIdx.x;
    const int wave = t >> 5;
    const int lane = t & 31;
    const int hf   = lane >> 4;
    const int idx  = lane & 15;
    const int st   = blockIdx.x;                     // query tile 0..63
    const int bat  = blockIdx.y;
    const size_t qrow0 = (size_t)(bat * SEQ + st * 16);

    // ---- stage Q tile: aligned 8-byte loads ----
    {
        const int row = t >> 4, col = (t & 15) << 2;
        *reinterpret_cast<unsigned long long*>(&q_lds[row][col]) =
            *reinterpret_cast<const unsigned long long*>(
                q + (qrow0 + row) * QKD + col);
    }
    __syncthreads();

    // ---- phase 1: scores = Q . K^T  (each wave owns 8 key tiles) ----
    for (int it = 0; it < 8; ++it) {
        const int kt = wave * 8 + it;                // key tile 0..63
        v8f c = {};
#pragma unroll
        for (int ks = 0; ks < QKD; ks += 32) {
            Frag16 fa, fb;
            const size_t krow = (size_t)(bat * SEQ + kt * 16 + idx) * QKD;
#pragma unroll
            for (int j = 0; j < 8; ++j) {
                const int kb = wmma_kb(j, hf);
                fa.u[j] = *reinterpret_cast<const unsigned int*>(&q_lds[idx][ks + kb]);
                fb.u[j] = *reinterpret_cast<const unsigned int*>(&km[krow + ks + kb]);
            }
            c = __builtin_amdgcn_wmma_f32_16x16x32_bf16(
                    false, fa.v, false, fb.v, (short)0, c, false, false);
        }
#pragma unroll
        for (int r = 0; r < 8; ++r)
            p_lds[r + hf * 8][kt * 16 + idx] = (__bf16)c[r];
    }
    __syncthreads();

    // ---- phase 2: row softmax (256 thr = 16 rows x 16 chunks of 64) ----
    const int srow = t >> 4;
    const int schk = t & 15;
    {
        float m = -1e30f;
        for (int cdx = schk * 64; cdx < schk * 64 + 64; ++cdx)
            m = fmaxf(m, (float)p_lds[srow][cdx]);
        red[srow][schk] = m;
    }
    __syncthreads();
    if (t < 16) {
        float m = -1e30f;
        for (int i = 0; i < 16; ++i) m = fmaxf(m, red[t][i]);
        rowmax[t] = m;
    }
    __syncthreads();
    {
        const float m = rowmax[srow];
        float s = 0.0f;
        for (int cdx = schk * 64; cdx < schk * 64 + 64; ++cdx) {
            const float e = __expf((float)p_lds[srow][cdx] - m);
            s += e;
            p_lds[srow][cdx] = (__bf16)e;
        }
        red[srow][schk] = s;
    }
    __syncthreads();
    if (t < 16) {
        float s = 0.0f;
        for (int i = 0; i < 16; ++i) s += red[t][i];
        invsum[t] = 1.0f / s;
    }
    __syncthreads();

    // ---- phase 3: out = P . V  (68 column tiles over DPAD, per-wave) ----
    for (int it = 0; it < 9; ++it) {
        const int nt = wave + 8 * it;                // wave-uniform
        if (nt >= DPAD / 16) break;
        const int colg = nt * 16 + idx;
        // V^T: this lane's column of V is a contiguous row of vt
        const __bf16* vcol = vt + (size_t)colg * NB + (size_t)bat * SEQ;
        v8f c = {};
        for (int kk0 = 0; kk0 < SEQ; kk0 += 32) {
            Frag16 fa, fb;
#pragma unroll
            for (int j = 0; j < 8; ++j) {
                const int kb = wmma_kb(j, hf);
                fa.u[j] = *reinterpret_cast<const unsigned int*>(&p_lds[idx][kk0 + kb]);
                fb.u[j] = *reinterpret_cast<const unsigned int*>(vcol + kk0 + kb);
            }
            c = __builtin_amdgcn_wmma_f32_16x16x32_bf16(
                    false, fa.v, false, fb.v, (short)0, c, false, false);
        }
#pragma unroll
        for (int r = 0; r < 8; ++r) {
            const int rl = r + hf * 8;
            out[(qrow0 + rl) * DPAD + colg] = (__bf16)(c[r] * invsum[rl]);
        }
    }
}

// ---------------------------------------------------------------------------
// out[b] = sum_ij a3[b,i,j] * coeffs[i,j] + offset   (a3 has stride DPAD)
// ---------------------------------------------------------------------------
__global__ __launch_bounds__(256) void k_reduce(
        const __bf16* __restrict__ a3, const float* __restrict__ coeffs,
        const float* __restrict__ offset, float* __restrict__ outp) {
    __shared__ float sred[256];
    const int b = blockIdx.x;
    const __bf16* base = a3 + (size_t)b * SEQ * DPAD;
    float s = 0.0f;
    for (int r = 0; r < SEQ; ++r) {
        const __bf16* arow = base + (size_t)r * DPAD;
        const float*  crow = coeffs + (size_t)r * DIM;
        for (int c = threadIdx.x; c < DIM; c += 256)
            s += (float)arow[c] * crow[c];
    }
    sred[threadIdx.x] = s;
    __syncthreads();
    for (int off = 128; off > 0; off >>= 1) {
        if (threadIdx.x < off) sred[threadIdx.x] += sred[threadIdx.x + off];
        __syncthreads();
    }
    if (threadIdx.x == 0) outp[b] = sred[0] + offset[0];
}

// ---------------------------------------------------------------------------
extern "C" void kernel_launch(void* const* d_in, const int* in_sizes, int n_in,
                              void* d_out, int out_size, void* d_ws, size_t ws_size,
                              hipStream_t stream) {
    (void)in_sizes; (void)n_in; (void)out_size; (void)ws_size;

    const float* x      = (const float*)d_in[0];
    const float* Wq[3]  = {(const float*)d_in[1], (const float*)d_in[4], (const float*)d_in[7]};
    const float* Wk[3]  = {(const float*)d_in[2], (const float*)d_in[5], (const float*)d_in[8]};
    const float* Wv[3]  = {(const float*)d_in[3], (const float*)d_in[6], (const float*)d_in[9]};
    const float* rnd    = (const float*)d_in[10];
    const float* coeffs = (const float*)d_in[11];
    const float* offset = (const float*)d_in[12];

    char* ws = (char*)d_ws;
    size_t off = 0;
    auto alloc = [&](size_t nbytes) -> void* {
        void* p = (void*)(ws + off);
        off += (nbytes + 255) & ~(size_t)255;
        return p;
    };

    const size_t nbp = (size_t)NB * DPAD;
    __bf16* xb   = (__bf16*)alloc(nbp * 2);                    // x, padded cols
    __bf16* rb   = (__bf16*)alloc((size_t)SEQ * SEQ * 2);      // 1024x1024
    __bf16 *wqb[3], *wkb[3], *wvb[3];
    for (int l = 0; l < 3; ++l) {
        wqb[l] = (__bf16*)alloc((size_t)DPAD * QKD * 2);       // K-padded rows
        wkb[l] = (__bf16*)alloc((size_t)DPAD * QKD * 2);
        wvb[l] = (__bf16*)alloc((size_t)DPAD * DPAD * 2);      // fully padded
    }
    __bf16* act0 = (__bf16*)alloc(nbp * 2);
    __bf16* act1 = (__bf16*)alloc(nbp * 2);
    __bf16* qb   = (__bf16*)alloc((size_t)NB * QKD * 2);
    __bf16* kb   = (__bf16*)alloc((size_t)NB * QKD * 2);
    __bf16* vtb  = (__bf16*)alloc(nbp * 2);                    // V^T [DPAD][NB]

    auto cvt = [&](const float* s, __bf16* d, int sr, int sc, int dr, int ds) {
        const long long n = (long long)dr * ds;
        k_cvt_pad<<<dim3((unsigned)((n + 255) / 256)), dim3(256), 0, stream>>>(
            s, d, sr, sc, dr, ds);
    };

    // bf16 staging with zero padding
    cvt(x, xb, NB, DIM, NB, DPAD);
    cvt(rnd, rb, SEQ, SEQ, SEQ, SEQ);
    for (int l = 0; l < 3; ++l) {
        cvt(Wq[l], wqb[l], DIM, QKD, DPAD, QKD);
        cvt(Wk[l], wkb[l], DIM, QKD, DPAD, QKD);
        cvt(Wv[l], wvb[l], DIM, DIM, DPAD, DPAD);
    }

    // y = x[:, :, :1024] @ R ; then col 1024 passthrough + zero pad cols
    k_gemm_bf16<<<dim3(NB / 64, SEQ / 64), 256, 0, stream>>>(
        xb, DPAD, rb, SEQ, act0, DPAD, SEQ, 0);
    {
        const long long n = (long long)NB * 64;
        k_fix_lastcols<<<dim3((unsigned)((n + 255) / 256)), dim3(256), 0, stream>>>(
            xb, act0);
    }

    __bf16* cur = act0;
    __bf16* nxt = act1;
    for (int l = 0; l < 3; ++l) {
        k_gemm_bf16<<<dim3(NB / 64, QKD / 64), 256, 0, stream>>>(
            cur, DPAD, wqb[l], QKD, qb, QKD, DPAD, 0);
        k_gemm_bf16<<<dim3(NB / 64, QKD / 64), 256, 0, stream>>>(
            cur, DPAD, wkb[l], QKD, kb, QKD, DPAD, 0);
        k_gemm_bf16<<<dim3(NB / 64, DPAD / 64), 256, 0, stream>>>(
            cur, DPAD, wvb[l], DPAD, vtb, NB, DPAD, 1);        // -> V^T
        k_attention<<<dim3(SEQ / 16, BATCH), 256, 0, stream>>>(qb, kb, vtb, nxt);
        __bf16* tmp = cur; cur = nxt; nxt = tmp;
    }

    k_reduce<<<dim3(BATCH), 256, 0, stream>>>(cur, coeffs, offset, (float*)d_out);
}
